// RNN_85890755985596
// MI455X (gfx1250) — matrix operand order repaired and checked by reading, not compiled, not run on hardware
//
#include <hip/hip_runtime.h>
#include <math.h>

// ---------------------------------------------------------------------------
// Vanilla tanh RNN for MI455X (gfx1250, wave32, WMMA).
//   B=64, T=2048, D=256, H=256
// Phase 1: xW = x @ W + b    -> written into d_out (same shape/size as output)
// Phase 2: h_t = tanh(xW_t + h_{t-1} @ W_rec), overwriting d_out in place.
// Matrix math: v_wmma_f32_16x16x32_bf16 (bf16 inputs, f32 accumulate).
// This revision uses a CDNA5 split workgroup barrier (s_barrier_signal -1 /
// s_barrier_wait -1) in the scan so the 8 global states-stores per timestep
// execute inside the signal->wait window instead of on the barrier-to-barrier
// critical path.
// ---------------------------------------------------------------------------

typedef __bf16 bf16_t;
typedef __attribute__((ext_vector_type(16))) __bf16 v16bf;
typedef __attribute__((ext_vector_type(8)))  __bf16 v8bf;
typedef __attribute__((ext_vector_type(8)))  float   v8f;

#define Bn 64
#define Tn 2048
#define Dn 256
#define Hn 256
#define LDPAD 8            // 16B row padding -> conflict-free ds_load_b128
#define LDW  (Hn + LDPAD)  // padded LDS row length in bf16 elements

// A-fragment (16x32 bf16) from an LDS tile stored row-major [16][LDW].
// Per ISA layout: lane l holds row (l&15); half = l>>4 selects
// K = half*8 + {0..7} and K = half*8 + {16..23}  -> two contiguous 16B loads.
__device__ __forceinline__ v16bf load_a_frag(const bf16_t* tile, int row,
                                             int half, int kk) {
  const bf16_t* p = tile + row * LDW + kk * 32 + half * 8;
  v8bf lo = *(const v8bf*)(p);
  v8bf hi = *(const v8bf*)(p + 16);
  return __builtin_shufflevector(lo, hi, 0, 1, 2, 3, 4, 5, 6, 7,
                                         8, 9, 10, 11, 12, 13, 14, 15);
}

// B-fragment (32x16 bf16) from a global f32 matrix [K][ncols] row-major.
// Lane l holds column (l&15); K = half*8 + {0..7} and half*8 + {16..23} for K-tile kk.
__device__ __forceinline__ v16bf load_b_frag(const float* __restrict__ w,
                                             int ncols, int col, int half,
                                             int kk) {
  v16bf b;
#pragma unroll
  for (int e = 0; e < 16; ++e) {
    int k = kk * 32 + half * 8 + e + (e >= 8 ? 8 : 0);
    b[e] = (bf16_t)w[(size_t)k * ncols + col];
  }
  return b;
}

// ---------------------------------------------------------------------------
// Phase 1: xW = x @ W + b   (bandwidth bound; 512 WGs x 16 waves)
// Each workgroup handles 256 rows of x (16 M-tiles); wave w owns N-tile w.
// W B-fragments stay in 64 VGPRs. x tiles staged in double-buffered LDS
// (bf16), one barrier per M-tile, with global reads pipelined one tile ahead.
// ---------------------------------------------------------------------------
__global__ __launch_bounds__(512) void rnn_xproj(const float* __restrict__ x,
                                                 const float* __restrict__ W,
                                                 const float* __restrict__ bias,
                                                 float* __restrict__ out) {
  __shared__ __align__(16) bf16_t xs[2][16 * LDW];
  const int lane = threadIdx.x & 31;
  const int wv   = threadIdx.x >> 5;   // 0..15 -> N-tile
  const int half = lane >> 4;
  const int l15  = lane & 15;
  const int col  = wv * 16 + l15;

  v16bf bw[8];
#pragma unroll
  for (int kk = 0; kk < 8; ++kk) bw[kk] = load_b_frag(W, Hn, col, half, kk);
  const float bn = bias[col];

  const int row_base = blockIdx.x * 256;
  // Each thread owns 8 fixed (row, col) slots of the 16x256 x tile:
  // e = threadIdx.x + j*512  ->  r = e>>8, cc = e&255.
  float xr[8];
#pragma unroll
  for (int j = 0; j < 8; ++j) {
    int e = threadIdx.x + j * 512;
    xr[j] = x[(size_t)(row_base + (e >> 8)) * Dn + (e & 255)];
  }

  for (int mt = 0; mt < 16; ++mt) {
    bf16_t* buf = xs[mt & 1];
    const int row0 = row_base + mt * 16;
    // Commit prefetched x tile (bf16) to LDS.
#pragma unroll
    for (int j = 0; j < 8; ++j) {
      int e = threadIdx.x + j * 512;
      buf[(e >> 8) * LDW + (e & 255)] = (bf16_t)xr[j];
    }
    // Issue next tile's global reads now; they overlap barrier + WMMAs.
    if (mt + 1 < 16) {
#pragma unroll
      for (int j = 0; j < 8; ++j) {
        int e = threadIdx.x + j * 512;
        xr[j] = x[(size_t)(row0 + 16 + (e >> 8)) * Dn + (e & 255)];
      }
    }
    __syncthreads();  // single barrier/iter is safe with double buffering

    v8f acc = {};
    v16bf a_cur = load_a_frag(buf, l15, half, 0);
#pragma unroll
    for (int kk = 0; kk < 8; ++kk) {
      v16bf a_nxt = a_cur;
      if (kk < 7) a_nxt = load_a_frag(buf, l15, half, kk + 1);  // in flight
      acc = __builtin_amdgcn_wmma_f32_16x16x32_bf16(false, a_cur, false,
                                                    bw[kk], (short)0, acc,
                                                    false, false);
      a_cur = a_nxt;
    }
    // C/D layout: VGPR r -> (m = r + half*8, n = l15); bias is per-column.
#pragma unroll
    for (int r = 0; r < 8; ++r)
      out[(size_t)(row0 + half * 8 + r) * Hn + col] = acc[r] + bn;
  }
}

// ---------------------------------------------------------------------------
// Phase 2: sequential scan. 4 workgroups x 16 waves; workgroup g owns batch
// rows [16g, 16g+16). W_rec B-fragments live in 64 VGPRs for all 2048 steps.
// h ping-pongs between two bf16 LDS buffers. Per step:
//   - prefetch xW(t+1) into the next accumulator (global, off critical path)
//   - 8 pipelined WMMAs (A fragments loaded one K-step ahead from LDS)
//   - tanh, publish h(t+1) to LDS, s_wait_dscnt 0, s_barrier_signal -1
//   - global states-stores happen inside the signal->wait window
//   - s_barrier_wait -1
// ---------------------------------------------------------------------------
__global__ __launch_bounds__(512) void rnn_scan(const float* __restrict__ Wrec,
                                                float* __restrict__ out) {
  __shared__ __align__(16) bf16_t hb[2][16 * LDW];
  const int lane = threadIdx.x & 31;
  const int wv   = threadIdx.x >> 5;
  const int half = lane >> 4;
  const int l15  = lane & 15;
  const int col  = wv * 16 + l15;
  const int b0   = blockIdx.x * 16;

  v16bf bw[8];
#pragma unroll
  for (int kk = 0; kk < 8; ++kk) bw[kk] = load_b_frag(Wrec, Hn, col, half, kk);

  // h0 = 0
  for (int e = threadIdx.x; e < 16 * LDW; e += 512) hb[0][e] = (bf16_t)0.f;
  __syncthreads();

  const size_t rstride = (size_t)Tn * Hn;  // row stride of [B][T][H]
  float* basep = out + (size_t)(b0 + half * 8) * rstride + col;

  // C init for t=0 = xW[:, 0, :]
  v8f c = {};
#pragma unroll
  for (int r = 0; r < 8; ++r) c[r] = basep[(size_t)r * rstride];

  int cur = 0;
  for (int t = 0; t < Tn; ++t) {
    // Prefetch next step's xW tile into registers (a full step of slack
    // before it is consumed as the t+1 accumulator).
    v8f cnext = {};
    if (t + 1 < Tn) {
#pragma unroll
      for (int r = 0; r < 8; ++r)
        cnext[r] = basep[(size_t)r * rstride + (size_t)(t + 1) * Hn];
    }

    // Pipelined A-fragment loads: K-step kk+1's ds_load_b128 pair is issued
    // before the WMMA for kk, so the dscnt wait overlaps WMMA execution.
    v16bf a_cur = load_a_frag(hb[cur], l15, half, 0);
#pragma unroll
    for (int kk = 0; kk < 8; ++kk) {
      v16bf a_nxt = a_cur;
      if (kk < 7) a_nxt = load_a_frag(hb[cur], l15, half, kk + 1);
      c = __builtin_amdgcn_wmma_f32_16x16x32_bf16(false, a_cur, false, bw[kk],
                                                  (short)0, c, false, false);
      a_cur = a_nxt;
    }

    // tanh and publish h(t+1) to the other LDS buffer first...
    float v[8];
    bf16_t* hn = hb[cur ^ 1];
#pragma unroll
    for (int r = 0; r < 8; ++r) {
      v[r] = tanhf(c[r]);
      hn[(half * 8 + r) * LDW + col] = (bf16_t)v[r];
    }
    // ...make the h stores visible, then signal arrival. All of this wave's
    // hb[cur] A-fragment loads were consumed (dscnt-ordered) above, so the
    // WAR on the ping-pong buffer is safe with one signal+wait per step.
    asm volatile("s_wait_dscnt 0x0" ::: "memory");
    asm volatile("s_barrier_signal -1" ::: "memory");

    // Global states-stores ride inside the signal->wait window, off the
    // inter-wave critical path.
#pragma unroll
    for (int r = 0; r < 8; ++r)
      basep[(size_t)r * rstride + (size_t)t * Hn] = v[r];     // states[b,t,:]

    asm volatile("s_barrier_wait -1" ::: "memory");
    cur ^= 1;
    c = cnext;
  }
}

// ---------------------------------------------------------------------------
extern "C" void kernel_launch(void* const* d_in, const int* in_sizes, int n_in,
                              void* d_out, int out_size, void* d_ws,
                              size_t ws_size, hipStream_t stream) {
  const float* x    = (const float*)d_in[0];  // [B,T,D]
  const float* W    = (const float*)d_in[1];  // [D,H]
  const float* Wrec = (const float*)d_in[2];  // [H,H]
  const float* b    = (const float*)d_in[3];  // [H]
  float* out = (float*)d_out;                 // [B,T,H]

  // Phase 1: 512 WGs x 512 threads, each WG does 256 rows of the BT=131072.
  rnn_xproj<<<dim3((Bn * Tn) / 256), dim3(512), 0, stream>>>(x, W, b, out);
  // Phase 2: 4 WGs x 512 threads (16 batch rows each), sequential over T.
  rnn_scan<<<dim3(Bn / 16), dim3(512), 0, stream>>>(Wrec, out);
}
